// CloudSense_51634096832520
// MI455X (gfx1250) — compile-verified, matrix operand order checked by reading, not run on hardware
//
#include <hip/hip_runtime.h>

// ---------------------------------------------------------------------------
// VQ-VAE codebook quantization for MI455X (gfx1250, wave32, WMMA).
// distances = ||z||^2 - 2 z.e^T + ||e||^2 ; argmin over 4096 codes; gather.
// GEMM (137 GFLOP) on V_WMMA_F32_16X16X32_BF16 with hi/lo bf16 split (3 wmmas
// per k-step ~ fp32 accuracy). Embedding chunks are double-buffered in LDS
// via GLOBAL_LOAD_ASYNC_TO_LDS_B128 (ASYNCcnt) to overlap L2 fetch w/ WMMA.
// ---------------------------------------------------------------------------

#define N_TOKENS 65536
#define LATENT_D 256
#define N_CODES  4096

#define M_TILE   64           // tokens per workgroup (4 waves x 16 rows)
#define N_CHUNK  16           // codes per inner iteration (one wmma N-tile)
#define THREADS  128          // 4 wave32
#define N_ITERS  (N_CODES / N_CHUNK)

typedef __attribute__((ext_vector_type(16))) __bf16         v16bf;
typedef __attribute__((ext_vector_type(8)))  float          v8f;
typedef __attribute__((ext_vector_type(4)))  float          f32x4;
typedef __attribute__((ext_vector_type(4)))  unsigned int   u32x4;
typedef __attribute__((ext_vector_type(4)))  unsigned short u16x4;
typedef __attribute__((__vector_size__(16))) int            b128v;  // builtin pointee

union ABFrag { f32x4 f4[2]; v16bf v; };   // 32 bytes: one wmma A/B operand

#if defined(__has_builtin)
#  if __has_builtin(__builtin_amdgcn_global_load_async_to_lds_b128)
#    define HAVE_ASYNC_LDS 1
#  endif
#endif
#ifndef HAVE_ASYNC_LDS
#  define HAVE_ASYNC_LDS 0
#endif

__device__ __forceinline__ unsigned short f32_to_bf16_rne(float x) {
  unsigned u = __builtin_bit_cast(unsigned, x);
  unsigned r = u + 0x7FFFu + ((u >> 16) & 1u);   // round-to-nearest-even
  return (unsigned short)(r >> 16);
}
__device__ __forceinline__ float bf16_bits_to_f32(unsigned short h) {
  unsigned u = ((unsigned)h) << 16;
  return __builtin_bit_cast(float, u);
}

// 16-byte global -> LDS copy; async DMA path on CDNA5 (no VGPR round-trip).
__device__ __forceinline__ void copy_b128_to_lds(void* lds, const void* g) {
#if HAVE_ASYNC_LDS
  __builtin_amdgcn_global_load_async_to_lds_b128(
      (__attribute__((address_space(1))) b128v*)g,
      (__attribute__((address_space(3))) b128v*)lds, 0, 0);
#else
  *(u32x4*)lds = *(const u32x4*)g;
#endif
}

__device__ __forceinline__ void wait_async_copies() {
#if HAVE_ASYNC_LDS
#  if __has_builtin(__builtin_amdgcn_s_wait_asynccnt)
  __builtin_amdgcn_s_wait_asynccnt(0);
#  else
  asm volatile("s_wait_asynccnt 0x0" ::: "memory");
#  endif
#endif
}

// ---------------------------------------------------------------------------
// Prep: split embedding into bf16 hi/lo planes (workspace) + fp32 ||e||^2.
// ---------------------------------------------------------------------------
__global__ __launch_bounds__(256)
void vq_prep_kernel(const float* __restrict__ emb,
                    unsigned short* __restrict__ e_hi,
                    unsigned short* __restrict__ e_lo,
                    float* __restrict__ e_norm) {
  const int k = blockIdx.x;        // 0..4095
  const int t = threadIdx.x;       // 0..255
  const float x = emb[(size_t)k * LATENT_D + t];
  const unsigned short h = f32_to_bf16_rne(x);
  const float lo = x - bf16_bits_to_f32(h);
  e_hi[(size_t)k * LATENT_D + t] = h;
  e_lo[(size_t)k * LATENT_D + t] = f32_to_bf16_rne(lo);

  __shared__ float red[256];
  red[t] = x * x;
  __syncthreads();
  for (int s = 128; s > 0; s >>= 1) {
    if (t < s) red[t] += red[t + s];
    __syncthreads();
  }
  if (t == 0) e_norm[k] = red[0];
}

// ---------------------------------------------------------------------------
// Fused distance + argmin + gather. One block per 64 tokens.
// LDS ~96 KB -> 3 workgroups per WGP (12 waves).
// ---------------------------------------------------------------------------
__global__ __launch_bounds__(THREADS)
void vq_main_kernel(const float* __restrict__ z,
                    const float* __restrict__ emb,
                    const unsigned short* __restrict__ e_hi,
                    const unsigned short* __restrict__ e_lo,
                    const float* __restrict__ e_norm,
                    float* __restrict__ out_q,
                    float* __restrict__ out_idx) {
  __shared__ unsigned short s_zh[M_TILE * LATENT_D];          // 32 KB
  __shared__ unsigned short s_zl[M_TILE * LATENT_D];          // 32 KB
  __shared__ unsigned short s_eh[2][N_CHUNK * LATENT_D];      // 16 KB (dbuf)
  __shared__ unsigned short s_el[2][N_CHUNK * LATENT_D];      // 16 KB (dbuf)
  __shared__ int s_idx[M_TILE];

  const int tid    = threadIdx.x;
  const int wave   = tid >> 5;     // 0..3 -> 16-row M slice
  const int lane   = tid & 31;
  const int half   = lane >> 4;    // 0: lanes 0-15, 1: lanes 16-31
  const int l16    = lane & 15;
  const int block0 = blockIdx.x * M_TILE;

  // ---- stage z tile, splitting fp32 -> bf16 hi + lo (coalesced b128 loads)
  {
    const f32x4* z4 = (const f32x4*)(z + (size_t)block0 * LATENT_D);
    for (int i = tid; i < M_TILE * LATENT_D / 4; i += THREADS) {
      f32x4 v = z4[i];
      u16x4 h, l;
      #pragma unroll
      for (int c = 0; c < 4; ++c) {
        unsigned short hb = f32_to_bf16_rne(v[c]);
        h[c] = hb;
        l[c] = f32_to_bf16_rne(v[c] - bf16_bits_to_f32(hb));
      }
      ((u16x4*)s_zh)[i] = h;
      ((u16x4*)s_zl)[i] = l;
    }
  }

  // ---- async-stage one 16-code hi/lo chunk (2 x 8 KB) into buffer `buf`
  auto stage_chunk = [&](int buf, int n0) {
    const char* sh = (const char*)(e_hi + (size_t)n0 * LATENT_D);
    const char* sl = (const char*)(e_lo + (size_t)n0 * LATENT_D);
    char* dh = (char*)s_eh[buf];
    char* dl = (char*)s_el[buf];
    #pragma unroll
    for (int j = 0; j < (N_CHUNK * LATENT_D * 2) / (16 * THREADS); ++j) {
      const int o = (tid + j * THREADS) * 16;
      copy_b128_to_lds(dh + o, sh + o);
      copy_b128_to_lds(dl + o, sl + o);
    }
  };

  stage_chunk(0, 0);   // prologue: chunk 0 in flight

  float best[8];
  int   bidx[8];
  #pragma unroll
  for (int r = 0; r < 8; ++r) { best[r] = 3.4e38f; bidx[r] = 0x7fffffff; }

  for (int i = 0; i < N_ITERS; ++i) {
    const int cur = i & 1;
    wait_async_copies();   // own chunk-i copies landed in LDS
    __syncthreads();       // everyone's copies landed; chunk i-1 fully read
    if (i + 1 < N_ITERS) stage_chunk(cur ^ 1, (i + 1) * N_CHUNK);

    const unsigned short* eh = s_eh[cur];
    const unsigned short* el = s_el[cur];

    // ---- 16x16 tile of z.e^T, D=256 as 8 k-steps of 32
    v8f acc = {0.f, 0.f, 0.f, 0.f, 0.f, 0.f, 0.f, 0.f};
    #pragma unroll
    for (int ks = 0; ks < LATENT_D / 32; ++ks) {
      const int k0 = ks * 32;
      // ISA 7.12.2: 16-bit A 16x32 — lane<16 row M=l16 holds K {k0..k0+7,
      // k0+16..k0+23}; lane>=16 holds K {+8..+15, +24..+31}.
      const int za = (wave * 16 + l16) * LATENT_D + k0 + half * 8;
      const int ba = l16 * LATENT_D + k0 + half * 8;   // B col N = code row
      ABFrag ah, al, bh, bl;
      ah.f4[0] = *(const f32x4*)(s_zh + za);
      ah.f4[1] = *(const f32x4*)(s_zh + za + 16);
      al.f4[0] = *(const f32x4*)(s_zl + za);
      al.f4[1] = *(const f32x4*)(s_zl + za + 16);
      bh.f4[0] = *(const f32x4*)(eh + ba);
      bh.f4[1] = *(const f32x4*)(eh + ba + 16);
      bl.f4[0] = *(const f32x4*)(el + ba);
      bl.f4[1] = *(const f32x4*)(el + ba + 16);
      // z.e ~= zh.eh + zh.el + zl.eh  (drop zl.el, ~2^-16)
      acc = __builtin_amdgcn_wmma_f32_16x16x32_bf16(false, ah.v, false, bh.v,
                                                    (short)0, acc, false, false);
      acc = __builtin_amdgcn_wmma_f32_16x16x32_bf16(false, ah.v, false, bl.v,
                                                    (short)0, acc, false, false);
      acc = __builtin_amdgcn_wmma_f32_16x16x32_bf16(false, al.v, false, bh.v,
                                                    (short)0, acc, false, false);
    }

    // dist(m,n) = ||e_n||^2 - 2 z_m.e_n   (+||z_m||^2 is row-constant)
    const int   n0   = i * N_CHUNK;
    const float en   = e_norm[n0 + l16];
    const int   nidx = n0 + l16;
    #pragma unroll
    for (int r = 0; r < 8; ++r) {
      float d = en - 2.0f * acc[r];
      if (d < best[r]) { best[r] = d; bidx[r] = nidx; }
    }
  }

  // ---- argmin across the 16-lane N-groups (butterfly keeps halves separate)
  #pragma unroll
  for (int r = 0; r < 8; ++r) {
    float bv = best[r];
    int   bi = bidx[r];
    #pragma unroll
    for (int s = 1; s < 16; s <<= 1) {
      float ov = __shfl_xor(bv, s, 32);
      int   oi = __shfl_xor(bi, s, 32);
      if (ov < bv || (ov == bv && oi < bi)) { bv = ov; bi = oi; }
    }
    // acc VGPR r: lanes 0-15 row M=r, lanes 16-31 row M=r+8
    if (l16 == 0) s_idx[wave * 16 + half * 8 + r] = bi;
  }
  __syncthreads();

  // ---- gather winning fp32 codebook rows: 64 rows x 64 float4
  {
    const f32x4* e4 = (const f32x4*)emb;
    f32x4*       o4 = (f32x4*)out_q;
    for (int t = tid; t < M_TILE * (LATENT_D / 4); t += THREADS) {
      int row = t >> 6;
      int c   = t & 63;
      int kk  = s_idx[row];
      o4[(size_t)(block0 + row) * (LATENT_D / 4) + c] =
          e4[(size_t)kk * (LATENT_D / 4) + c];
    }
  }
  if (tid < M_TILE) out_idx[block0 + tid] = (float)s_idx[tid];
}

// ---------------------------------------------------------------------------
extern "C" void kernel_launch(void* const* d_in, const int* in_sizes, int n_in,
                              void* d_out, int out_size, void* d_ws, size_t ws_size,
                              hipStream_t stream) {
  (void)in_sizes; (void)n_in; (void)out_size; (void)ws_size;
  const float* z   = (const float*)d_in[0];   // [65536, 256]
  const float* emb = (const float*)d_in[1];   // [4096, 256]

  // workspace: e_hi (2 MB) | e_lo (2 MB) | e_norm (16 KB)
  char* ws = (char*)d_ws;
  unsigned short* e_hi   = (unsigned short*)ws;
  unsigned short* e_lo   = (unsigned short*)(ws + (size_t)2 * 1024 * 1024);
  float*          e_norm = (float*)(ws + (size_t)4 * 1024 * 1024);

  float* out_q   = (float*)d_out;                         // [65536, 256]
  float* out_idx = out_q + (size_t)N_TOKENS * LATENT_D;   // [65536]

  vq_prep_kernel<<<N_CODES, 256, 0, stream>>>(emb, e_hi, e_lo, e_norm);
  vq_main_kernel<<<N_TOKENS / M_TILE, THREADS, 0, stream>>>(
      z, emb, e_hi, e_lo, e_norm, out_q, out_idx);
}
